// LSTM_33157147525557
// MI455X (gfx1250) — matrix operand order repaired and checked by reading, not compiled, not run on hardware
//
#include <hip/hip_runtime.h>
#include <stdint.h>
#include <stddef.h>

// ---------------------------------------------------------------------------
// hyperLSTM step for MI455X (gfx1250): all GEMMs via v_wmma_f32_16x16x32_bf16,
// LDS staging via GLOBAL_LOAD_ASYNC_TO_LDS_B128 double-buffering (1 barrier +
// 1 s_wait_asynccnt per k-step, branch-free steady-state loop).
// ---------------------------------------------------------------------------

typedef __bf16 bf16_t;
typedef bf16_t v16bf __attribute__((ext_vector_type(16)));
typedef float  v8f   __attribute__((ext_vector_type(8)));
typedef int    v4i   __attribute__((ext_vector_type(4)));

#define BB 2048
#define DD 1024
#define HH 1024
#define ZZ 256
#define EE 64

#if defined(__has_builtin)
#if __has_builtin(__builtin_amdgcn_global_load_async_to_lds_b128) && \
    __has_builtin(__builtin_amdgcn_s_wait_asynccnt)
#define USE_ASYNC_LDS 1
#endif
#endif
#ifndef USE_ASYNC_LDS
#define USE_ASYNC_LDS 0
#endif

#if USE_ASYNC_LDS
typedef __attribute__((address_space(1))) v4i* as1_v4i_p;
typedef __attribute__((address_space(3))) v4i* as3_v4i_p;
#endif

#define TILE_ELEMS (128 * 32)   // one 128x32 bf16 tile

// ---------------- helpers ----------------

__device__ __forceinline__ bf16_t f2bf(float f) {
  union { float f; uint32_t u; } v; v.f = f;
  uint32_t r = v.u + 0x7FFFu + ((v.u >> 16) & 1u);   // round-to-nearest-even
  union { uint16_t s; bf16_t b; } o; o.s = (uint16_t)(r >> 16);
  return o.b;
}

__device__ __forceinline__ float sigf(float x) {
  return 1.0f / (1.0f + __expf(-x));
}

__device__ __forceinline__ v8f wmma_bf16(v16bf a, v16bf b, v8f c) {
  // v_wmma_f32_16x16x32_bf16: (neg_a, A, neg_b, B, c_mod, C, reuse_a, reuse_b)
  return __builtin_amdgcn_wmma_f32_16x16x32_bf16(false, a, false, b, (short)0, c,
                                                 false, false);
}

// A fragment (16x32 bf16) from LDS tile [128][32], row-major.
__device__ __forceinline__ v16bf load_a_frag(const bf16_t* As, int mbase, int lane) {
  const int row  = mbase + (lane & 15);
  const int half = lane >> 4;
  union { v16bf v; uint4 q[2]; } u;
  u.q[0] = *(const uint4*)(As + row * 32 + half * 16);
  u.q[1] = *(const uint4*)(As + row * 32 + half * 16 + 8);
  return u.v;
}

// B fragment (32x16 bf16) from LDS tile of B^T [128 rows=N][32 cols=K].
__device__ __forceinline__ v16bf load_b_frag(const bf16_t* Bts, int nbase, int lane) {
  const int n  = nbase + (lane & 15);
  const int kb = (lane >> 4) * 16;
  const uint4* p = (const uint4*)(Bts + n * 32 + kb);
  union { v16bf v; uint4 q[2]; } u;
  u.q[0] = p[0];
  u.q[1] = p[1];
  return u.v;
}

// one k-step of fragment loads + 8 WMMAs for this wave
__device__ __forceinline__ void wmma_step(v8f (&acc)[2][4],
                                          const bf16_t* Asb, const bf16_t* Btsb,
                                          int wm, int wn, int lane) {
  v16bf a0 = load_a_frag(Asb, wm + 0,  lane);
  v16bf a1 = load_a_frag(Asb, wm + 16, lane);
  v16bf b0 = load_b_frag(Btsb, wn + 0,  lane);
  v16bf b1 = load_b_frag(Btsb, wn + 16, lane);
  v16bf b2 = load_b_frag(Btsb, wn + 32, lane);
  v16bf b3 = load_b_frag(Btsb, wn + 48, lane);
  acc[0][0] = wmma_bf16(a0, b0, acc[0][0]);
  acc[0][1] = wmma_bf16(a0, b1, acc[0][1]);
  acc[0][2] = wmma_bf16(a0, b2, acc[0][2]);
  acc[0][3] = wmma_bf16(a0, b3, acc[0][3]);
  acc[1][0] = wmma_bf16(a1, b0, acc[1][0]);
  acc[1][1] = wmma_bf16(a1, b1, acc[1][1]);
  acc[1][2] = wmma_bf16(a1, b2, acc[1][2]);
  acc[1][3] = wmma_bf16(a1, b3, acc[1][3]);
}

#if USE_ASYNC_LDS
__device__ __forceinline__ void async_b128(const bf16_t* g, bf16_t* l) {
  __builtin_amdgcn_global_load_async_to_lds_b128((as1_v4i_p)g, (as3_v4i_p)l, 0, 0);
}

// Issue async global->LDS copies for one A tile + one Bt tile (4 x b128/thread).
__device__ __forceinline__ void issue_tiles_async(const bf16_t* A, int lda,
                                                  const bf16_t* Bt, int ldbt,
                                                  int k0, bf16_t* As, bf16_t* Bts,
                                                  int tid) {
  const int row = tid >> 1;
  const int col = (tid & 1) * 16;
  const bf16_t* ga = A + (size_t)row * lda + k0 + col;
  bf16_t* sa = As + row * 32 + col;
  async_b128(ga,     sa);
  async_b128(ga + 8, sa + 8);
  const bf16_t* gb = Bt + (size_t)row * ldbt + k0 + col;
  bf16_t* sb = Bts + row * 32 + col;
  async_b128(gb,     sb);
  async_b128(gb + 8, sb + 8);
}
#endif

// Accumulate C[128x128] += A[128xK] * Bt[128(N) x K]^T over K, 32 per step.
// As/Bts are double-buffered: 2 x (128*32) bf16 each. K must be >= 64 and a
// multiple of 32 (true for all call sites: 2304, 2048, 256, 64).
__device__ __forceinline__ void gemm_acc(v8f (&acc)[2][4],
                                         const bf16_t* __restrict__ A, int lda,
                                         const bf16_t* __restrict__ Bt, int ldbt,
                                         int K,
                                         bf16_t* As, bf16_t* Bts, int tid) {
  const int lane = tid & 31;
  const int wave = tid >> 5;
  const int wm = (wave & 3) * 32;   // wave M offset inside 128
  const int wn = (wave >> 2) * 64;  // wave N offset inside 128

#if USE_ASYNC_LDS
  issue_tiles_async(A, lda, Bt, ldbt, 0, As, Bts, tid);
  int buf = 0;
  // steady state: branch-free; one wait + one barrier per k-step
  for (int k0 = 0; k0 < K - 32; k0 += 32, buf ^= 1) {
    __builtin_amdgcn_s_wait_asynccnt(0);   // tile 'buf' resident
    __syncthreads();                        // publish; other buf fully consumed
    issue_tiles_async(A, lda, Bt, ldbt, k0 + 32,
                      As + (buf ^ 1) * TILE_ELEMS, Bts + (buf ^ 1) * TILE_ELEMS,
                      tid);                 // overlap copy with compute below
    wmma_step(acc, As + buf * TILE_ELEMS, Bts + buf * TILE_ELEMS, wm, wn, lane);
  }
  // peeled last step
  __builtin_amdgcn_s_wait_asynccnt(0);
  __syncthreads();
  wmma_step(acc, As + buf * TILE_ELEMS, Bts + buf * TILE_ELEMS, wm, wn, lane);
  __syncthreads();   // LDS safe for the next gemm_acc call
#else
  for (int k0 = 0; k0 < K; k0 += 32) {
    __syncthreads();
    {
      const int row = tid >> 1;
      const int col = (tid & 1) * 16;
      const uint4* ga = (const uint4*)(A + (size_t)row * lda + k0 + col);
      uint4 a0 = ga[0], a1 = ga[1];
      const uint4* gb = (const uint4*)(Bt + (size_t)row * ldbt + k0 + col);
      uint4 b0 = gb[0], b1 = gb[1];
      uint4* sa = (uint4*)(As + row * 32 + col);
      sa[0] = a0; sa[1] = a1;
      uint4* sb = (uint4*)(Bts + row * 32 + col);
      sb[0] = b0; sb[1] = b1;
    }
    __syncthreads();
    wmma_step(acc, As, Bts, wm, wn, lane);
  }
#endif
}

__device__ __forceinline__ void zero_acc(v8f (&acc)[2][4]) {
#pragma unroll
  for (int i = 0; i < 2; ++i)
#pragma unroll
    for (int j = 0; j < 4; ++j)
#pragma unroll
      for (int r = 0; r < 8; ++r) acc[i][j][r] = 0.0f;
}

// ---------------- prep (convert / transpose to bf16) ----------------

__global__ void build_hcat_kernel(const float* __restrict__ hhat0,
                                  const float* __restrict__ h0,
                                  const float* __restrict__ x,
                                  bf16_t* __restrict__ hcat) {
  int idx = blockIdx.x * blockDim.x + threadIdx.x;
  if (idx >= BB * (ZZ + HH + DD)) return;
  int b = idx / (ZZ + HH + DD);
  int c = idx % (ZZ + HH + DD);
  float v;
  if (c < ZZ)            v = hhat0[b * ZZ + c];
  else if (c < ZZ + HH)  v = h0[b * HH + (c - ZZ)];
  else                   v = x[b * DD + (c - ZZ - HH)];
  hcat[idx] = f2bf(v);
}

// src [K,N] f32 row-major -> dst [N,K] bf16 row-major (B^T for WMMA)
__global__ void transpose_bf16_kernel(const float* __restrict__ src,
                                      bf16_t* __restrict__ dst, int K, int N) {
  int idx = blockIdx.x * blockDim.x + threadIdx.x;
  if (idx >= K * N) return;
  int n = idx / K;
  int k = idx % K;
  dst[idx] = f2bf(src[(size_t)k * N + n]);
}

__global__ void build_zwT_kernel(const float* __restrict__ zh,
                                 const float* __restrict__ zx,
                                 const float* __restrict__ zb,
                                 bf16_t* __restrict__ zwT) {
  int idx = blockIdx.x * blockDim.x + threadIdx.x;
  if (idx >= 768 * ZZ) return;
  int n = idx / ZZ;       // 0..767
  int k = idx % ZZ;       // 0..255
  int sel = n / 256;
  int nn = n % 256;
  const float* s = (sel == 0) ? zh : (sel == 1) ? zx : zb;
  zwT[idx] = f2bf(s[(size_t)k * 256 + nn]);
}

// dwT [3][4][1024][64] : dwT[s][g][h][e] = dweight_s[g][e][h]
__global__ void build_dwT_kernel(const float* __restrict__ dh,
                                 const float* __restrict__ dx,
                                 const float* __restrict__ db,
                                 bf16_t* __restrict__ dwT) {
  int idx = blockIdx.x * blockDim.x + threadIdx.x;
  if (idx >= 3 * 4 * HH * EE) return;
  int e = idx % EE;
  int h = (idx / EE) % HH;
  int g = (idx / (EE * HH)) % 4;
  int s = idx / (EE * HH * 4);
  const float* p = (s == 0) ? dh : (s == 1) ? dx : db;
  dwT[idx] = f2bf(p[((size_t)g * EE + e) * HH + h]);
}

__global__ void build_zbias_kernel(const float* __restrict__ zbh,
                                   const float* __restrict__ zbx,
                                   float* __restrict__ zbias) {
  int i = blockIdx.x * blockDim.x + threadIdx.x;
  if (i >= 768) return;
  zbias[i] = (i < 256) ? zbh[i] : (i < 512) ? zbx[i - 256] : 0.0f;
}

// ---------------- generic WMMA GEMM: C = A @ Bt^T (+bias) ----------------

__global__ __launch_bounds__(256) void wmma_gemm_kernel(
    const bf16_t* __restrict__ A, int lda,
    const bf16_t* __restrict__ Bt, int ldbt,
    const float* __restrict__ bias,
    float* __restrict__ Cf, bf16_t* __restrict__ Cbf,
    int ldc, int K) {
  __shared__ __align__(16) bf16_t As[2 * TILE_ELEMS];
  __shared__ __align__(16) bf16_t Bts[2 * TILE_ELEMS];
  const int tid = threadIdx.x;
  const int lane = tid & 31;
  const int wave = tid >> 5;
  const int bm = blockIdx.y * 128;
  const int bn = blockIdx.x * 128;

  v8f acc[2][4];
  zero_acc(acc);
  gemm_acc(acc, A + (size_t)bm * lda, lda, Bt + (size_t)bn * ldbt, ldbt, K,
           As, Bts, tid);

  const int wm = (wave & 3) * 32;
  const int wn = (wave >> 2) * 64;
  const int half = lane >> 4;
#pragma unroll
  for (int i = 0; i < 2; ++i) {
#pragma unroll
    for (int j = 0; j < 4; ++j) {
      const int col = bn + wn + j * 16 + (lane & 15);
      const float bb = bias ? bias[col] : 0.0f;
#pragma unroll
      for (int r = 0; r < 8; ++r) {
        const int row = bm + wm + i * 16 + half * 8 + r;
        const float v = acc[i][j][r] + bb;
        if (Cf)  Cf[(size_t)row * ldc + col] = v;
        if (Cbf) Cbf[(size_t)row * ldc + col] = f2bf(v);
      }
    }
  }
}

// ---------------- hyper-cell activation ----------------

__global__ void hyper_act_kernel(const float* __restrict__ hg,   // [B, 4Z]
                                 const float* __restrict__ chat0,
                                 float* __restrict__ out_hhat1,
                                 float* __restrict__ out_chat1,
                                 bf16_t* __restrict__ hhat1_bf) {
  int idx = blockIdx.x * blockDim.x + threadIdx.x;
  if (idx >= BB * ZZ) return;
  int b = idx / ZZ;
  int z = idx % ZZ;
  const float* row = hg + (size_t)b * (4 * ZZ);
  float i_h = row[z];
  float g_h = row[ZZ + z];
  float f_h = row[2 * ZZ + z];
  float o_h = row[3 * ZZ + z];
  float chat1 = sigf(f_h) * chat0[idx] + sigf(i_h) * tanhf(g_h);
  float hhat1 = sigf(o_h) * tanhf(chat1);
  out_chat1[idx] = chat1;
  out_hhat1[idx] = hhat1;
  hhat1_bf[idx] = f2bf(hhat1);
}

// ---------------- fused main-gates kernel ----------------
// gates[b, g*1024+h] = (hx@W) * d_yh * d_yx + d_yb + bias
__global__ __launch_bounds__(256) void gates_kernel(
    const bf16_t* __restrict__ hx,    // hcat + Z, lda = 2304, [B, 2048]
    const bf16_t* __restrict__ wT,    // [4096][2048]
    const bf16_t* __restrict__ zall,  // [B][768] (z_h | z_x | z_b)
    const bf16_t* __restrict__ dwT,   // [3][4][1024][64]
    const float* __restrict__ bias4h, // [4096]
    float* __restrict__ gates_out) {  // [B][4096]
  __shared__ __align__(16) bf16_t As[2 * TILE_ELEMS];
  __shared__ __align__(16) bf16_t Bts[2 * TILE_ELEMS];
  const int tid = threadIdx.x;
  const int lane = tid & 31;
  const int wave = tid >> 5;
  const int bm = blockIdx.y * 128;
  const int c0 = blockIdx.x * 128;   // stays within one gate (1024/128 = 8)
  const int g  = c0 >> 10;
  const int h0 = c0 & 1023;

  v8f accM[2][4];
  zero_acc(accM);
  gemm_acc(accM, hx + (size_t)bm * 2304, 2304,
           wT + (size_t)c0 * 2048, 2048, 2048, As, Bts, tid);

  const bf16_t* zrow = zall + (size_t)bm * 768 + g * EE;
  v8f tmp[2][4];

  // d_yh
  zero_acc(tmp);
  gemm_acc(tmp, zrow, 768,
           dwT + ((size_t)(0 * 4 + g) * 1024 + h0) * 64, 64, 64, As, Bts, tid);
#pragma unroll
  for (int i = 0; i < 2; ++i)
#pragma unroll
    for (int j = 0; j < 4; ++j) accM[i][j] *= tmp[i][j];

  // d_yx
  zero_acc(tmp);
  gemm_acc(tmp, zrow + 256, 768,
           dwT + ((size_t)(1 * 4 + g) * 1024 + h0) * 64, 64, 64, As, Bts, tid);
#pragma unroll
  for (int i = 0; i < 2; ++i)
#pragma unroll
    for (int j = 0; j < 4; ++j) accM[i][j] *= tmp[i][j];

  // d_yb
  zero_acc(tmp);
  gemm_acc(tmp, zrow + 512, 768,
           dwT + ((size_t)(2 * 4 + g) * 1024 + h0) * 64, 64, 64, As, Bts, tid);
#pragma unroll
  for (int i = 0; i < 2; ++i)
#pragma unroll
    for (int j = 0; j < 4; ++j) accM[i][j] += tmp[i][j];

  const int wm = (wave & 3) * 32;
  const int wn = (wave >> 2) * 64;
  const int half = lane >> 4;
#pragma unroll
  for (int i = 0; i < 2; ++i) {
#pragma unroll
    for (int j = 0; j < 4; ++j) {
      const int col = c0 + wn + j * 16 + (lane & 15);
      const float bb = bias4h[col];
#pragma unroll
      for (int r = 0; r < 8; ++r) {
        const int row = bm + wm + i * 16 + half * 8 + r;
        gates_out[(size_t)row * 4096 + col] = accM[i][j][r] + bb;
      }
    }
  }
}

// ---------------- final LSTM activation ----------------

__global__ void final_act_kernel(const float* __restrict__ gates, // [B][4096]
                                 const float* __restrict__ c0,
                                 float* __restrict__ out_h1,
                                 float* __restrict__ out_c1) {
  int idx = blockIdx.x * blockDim.x + threadIdx.x;
  if (idx >= BB * HH) return;
  int b = idx / HH;
  int h = idx % HH;
  const float* row = gates + (size_t)b * (4 * HH);
  float i_g = row[h];
  float g_g = row[HH + h];
  float f_g = row[2 * HH + h];
  float o_g = row[3 * HH + h];
  float c1 = sigf(f_g) * c0[idx] + sigf(i_g) * tanhf(g_g);
  float h1 = sigf(o_g) * tanhf(c1);
  out_c1[idx] = c1;
  out_h1[idx] = h1;
}

// ---------------- host launch ----------------

extern "C" void kernel_launch(void* const* d_in, const int* in_sizes, int n_in,
                              void* d_out, int out_size, void* d_ws, size_t ws_size,
                              hipStream_t stream) {
  (void)in_sizes; (void)n_in; (void)out_size; (void)ws_size;

  const float* x        = (const float*)d_in[0];
  const float* h0       = (const float*)d_in[1];
  const float* c0       = (const float*)d_in[2];
  const float* hhat0    = (const float*)d_in[3];
  const float* chat0    = (const float*)d_in[4];
  const float* hweight  = (const float*)d_in[5];   // [2304, 1024]
  const float* hbias    = (const float*)d_in[6];   // [1024]
  const float* zw_h     = (const float*)d_in[7];   // [256, 256]
  const float* zw_x     = (const float*)d_in[8];
  const float* zw_b     = (const float*)d_in[9];
  const float* zbias_h  = (const float*)d_in[10];  // [256]
  const float* zbias_x  = (const float*)d_in[11];
  const float* dw_h     = (const float*)d_in[12];  // [4, 64, 1024]
  const float* dw_x     = (const float*)d_in[13];
  const float* dw_b     = (const float*)d_in[14];
  const float* weight   = (const float*)d_in[15];  // [2048, 4096]
  const float* bias     = (const float*)d_in[16];  // [4096]

  float* out = (float*)d_out;
  float* out_h1    = out;
  float* out_c1    = out + (size_t)BB * HH;
  float* out_hhat1 = out + (size_t)2 * BB * HH;
  float* out_chat1 = out + (size_t)2 * BB * HH + (size_t)BB * ZZ;

  // workspace layout (byte offsets, 256-aligned)
  char* ws = (char*)d_ws;
  bf16_t* hcat_bf  = (bf16_t*)(ws + 0);          //  9,437,184  [2048][2304]
  bf16_t* hwT_bf   = (bf16_t*)(ws + 9437184);    //  4,718,592  [1024][2304]
  bf16_t* wT_bf    = (bf16_t*)(ws + 14155776);   // 16,777,216  [4096][2048]
  bf16_t* zwT_bf   = (bf16_t*)(ws + 30932992);   //    393,216  [768][256]
  bf16_t* dwT_bf   = (bf16_t*)(ws + 31326208);   //  1,572,864  [3][4][1024][64]
  float*  zbias    = (float*) (ws + 32899072);   //      3,072  [768]
  float*  hg       = (float*) (ws + 32902144);   //  8,388,608  [2048][1024]
  bf16_t* hhat1_bf = (bf16_t*)(ws + 41290752);   //  1,048,576  [2048][256]
  bf16_t* z_bf     = (bf16_t*)(ws + 42339328);   //  3,145,728  [2048][768]
  float*  gates    = (float*) (ws + 45485056);   // 33,554,432  [2048][4096]

  const int T = 256;
  auto nb = [](int n, int t) { return (n + t - 1) / t; };

  // --- prep: bf16 conversions / transposes ---
  build_hcat_kernel<<<nb(BB * (ZZ + HH + DD), T), T, 0, stream>>>(hhat0, h0, x, hcat_bf);
  transpose_bf16_kernel<<<nb(2304 * 1024, T), T, 0, stream>>>(hweight, hwT_bf, 2304, 1024);
  transpose_bf16_kernel<<<nb(2048 * 4096, T), T, 0, stream>>>(weight, wT_bf, 2048, 4096);
  build_zwT_kernel<<<nb(768 * 256, T), T, 0, stream>>>(zw_h, zw_x, zw_b, zwT_bf);
  build_dwT_kernel<<<nb(3 * 4 * HH * EE, T), T, 0, stream>>>(dw_h, dw_x, dw_b, dwT_bf);
  build_zbias_kernel<<<nb(768, T), T, 0, stream>>>(zbias_h, zbias_x, zbias);

  // --- hyper GEMM: hg = hcat @ hweight + hbias   [2048,2304]@[2304,1024] ---
  {
    dim3 grid(1024 / 128, 2048 / 128);
    wmma_gemm_kernel<<<grid, T, 0, stream>>>(hcat_bf, 2304, hwT_bf, 2304,
                                             hbias, hg, (bf16_t*)nullptr, 1024, 2304);
  }

  // --- hyper cell activations -> hhat1, chat1 (+ bf16 hhat1) ---
  hyper_act_kernel<<<nb(BB * ZZ, T), T, 0, stream>>>(hg, chat0, out_hhat1, out_chat1,
                                                     hhat1_bf);

  // --- z GEMM: z_all = hhat1 @ [zw_h|zw_x|zw_b] + [zb_h|zb_x|0] -> bf16 ---
  {
    dim3 grid(768 / 128, 2048 / 128);
    wmma_gemm_kernel<<<grid, T, 0, stream>>>(hhat1_bf, 256, zwT_bf, 256,
                                             zbias, (float*)nullptr, z_bf, 768, 256);
  }

  // --- fused main gates: (hx@W) * d_yh * d_yx + d_yb + bias ---
  {
    dim3 grid(4096 / 128, 2048 / 128);
    gates_kernel<<<grid, T, 0, stream>>>(hcat_bf + ZZ, wT_bf, z_bf, dwT_bf,
                                         bias, gates);
  }

  // --- final LSTM activation -> h1, c1 ---
  final_act_kernel<<<nb(BB * HH, T), T, 0, stream>>>(gates, c0, out_h1, out_c1);
}
